// clustering_dynamic_learning_common_center_45286135169475
// MI455X (gfx1250) — matrix operand order repaired and checked by reading, not compiled, not run on hardware
//
#include <hip/hip_runtime.h>
#include <hip/hip_bf16.h>

typedef __attribute__((ext_vector_type(16))) _Float16 v16h;
typedef __attribute__((ext_vector_type(8)))  float    v8f;
typedef __attribute__((ext_vector_type(4)))  float    v4f;
typedef __attribute__((ext_vector_type(4)))  unsigned int v4u;
typedef __attribute__((ext_vector_type(8)))  int      v8i;
typedef __attribute__((ext_vector_type(4)))  int      v4i;

#define B_ 32
#define N_ 500
#define K_ 32
#define C_ 16
#define D_ 96
#define S_ 96
#define ROWS_ (B_ * N_)   // 16000

// ---------------------------------------------------------------------------
// Kernel 1: per-n batchnorm stats over (B, D) -> scale a[n], shift c[n]
// ---------------------------------------------------------------------------
__global__ void k_bn_stats(const float* __restrict__ f,
                           const float* __restrict__ gamma,
                           const float* __restrict__ beta,
                           float* __restrict__ aS, float* __restrict__ cS) {
    __shared__ float s1[128], s2[128];
    const int n = blockIdx.x;
    const int t = threadIdx.x;
    float a1 = 0.f, a2 = 0.f;
    for (int i = t; i < B_ * D_; i += 128) {
        int b = i / D_, d = i - b * D_;
        float v = f[(size_t)(b * N_ + n) * D_ + d];
        a1 += v; a2 += v * v;
    }
    s1[t] = a1; s2[t] = a2;
    __syncthreads();
    for (int off = 64; off > 0; off >>= 1) {
        if (t < off) { s1[t] += s1[t + off]; s2[t] += s2[t + off]; }
        __syncthreads();
    }
    if (t == 0) {
        const float inv_cnt = 1.0f / (float)(B_ * D_);
        float mu  = s1[0] * inv_cnt;
        float var = s2[0] * inv_cnt - mu * mu;
        float inv = rsqrtf(var + 1e-5f);
        float g = gamma[n];
        aS[n] = inv * g;
        cS[n] = beta[n] - mu * inv * g;
    }
}

// ---------------------------------------------------------------------------
// Kernel 2: cw[c][d] = centroids[c] @ w1[D:2D,:] + b1[d]  and
//           w1t[d][k] = (f16) w1[k][d]   (transposed f16 copy of w1_top)
// ---------------------------------------------------------------------------
__global__ void k_prep(const float* __restrict__ cent,
                       const float* __restrict__ w1,
                       const float* __restrict__ b1,
                       float* __restrict__ cw,
                       _Float16* __restrict__ w1t) {
    int tid = blockIdx.x * blockDim.x + threadIdx.x;   // 1536 threads
    if (tid < C_ * D_) {
        int c = tid / D_, d = tid - c * D_;
        float acc = b1[d];
        for (int u = 0; u < D_; ++u)
            acc += cent[c * D_ + u] * w1[(D_ + u) * D_ + d];
        cw[tid] = acc;
    }
    for (int i = tid; i < D_ * D_; i += C_ * D_ * 6) {  // stride 9216 = D*D: one iter per thread
        int d = i / D_, k = i - d * D_;
        w1t[d * D_ + k] = (_Float16)w1[k * D_ + d];
    }
}

// ---------------------------------------------------------------------------
// Kernel 3: 16-row tile of fn @ w1_top via WMMA (f16 operands, f32 accum),
// then per-c relu/w2 reduction, softmax over C, store simi as f16.
// One wave32 per 16 rows; grid = 1000.
// ---------------------------------------------------------------------------
__global__ void k_score(const float* __restrict__ f,
                        const _Float16* __restrict__ w1t,
                        const float* __restrict__ w2,
                        const float* __restrict__ b2p,
                        const float* __restrict__ aS,
                        const float* __restrict__ cS,
                        const float* __restrict__ cw,
                        _Float16* __restrict__ simiH) {
    const int L = threadIdx.x;       // 0..31
    const int h = L >> 4;            // lane half
    const int m = L & 15;            // A row / B,D col
    const int r0 = blockIdx.x * 16;
    const int r  = r0 + m;           // flat b*N+n row
    const int n  = r % N_;
    const float a  = aS[n];
    const float c0 = cS[n];

    // A fragments (16x32 f16), 3 k-steps.  Element e = g*8+q maps to
    // k = t*32 + h*16 + g*8 + q  -> one contiguous 8-float run (two v4f) per (t,g).
    v16h A[3];
    const float* frow = f + (size_t)r * D_;
#pragma unroll
    for (int t = 0; t < 3; ++t) {
#pragma unroll
        for (int g = 0; g < 2; ++g) {
            const v4f* p = (const v4f*)(frow + t * 32 + h * 16 + g * 8);
            v4f lo = p[0], hi = p[1];
#pragma unroll
            for (int q = 0; q < 4; ++q) A[t][g * 8 + q]     = (_Float16)(lo[q] * a + c0);
#pragma unroll
            for (int q = 0; q < 4; ++q) A[t][g * 8 + 4 + q] = (_Float16)(hi[q] * a + c0);
        }
    }

    // 6 d-tiles of fnW1; B fragment = one contiguous 32B f16 load from w1t
    v8f acc[6];
#pragma unroll
    for (int nt = 0; nt < 6; ++nt) {
        v8f z;
#pragma unroll
        for (int i = 0; i < 8; ++i) z[i] = 0.f;
        const int dcol = nt * 16 + m;
#pragma unroll
        for (int t = 0; t < 3; ++t) {
            v16h Bf = *(const v16h*)(w1t + (size_t)dcol * D_ + t * 32 + h * 16);
            z = __builtin_amdgcn_wmma_f32_16x16x32_f16(
                    false, A[t], false, Bf, (short)0, z, false, false);
        }
        acc[nt] = z;
    }

    // s[row][c] = relu( sum_d relu(fnW1 + cw[c][d]) * w2[d] + b2 )
    __shared__ float s_lds[16][16];
    float w2c[6];
#pragma unroll
    for (int nt = 0; nt < 6; ++nt) w2c[nt] = w2[nt * 16 + m];
    const float b2v = b2p[0];

    for (int c = 0; c < 16; ++c) {
        float cwc[6];
#pragma unroll
        for (int nt = 0; nt < 6; ++nt) cwc[nt] = cw[c * D_ + nt * 16 + m];
#pragma unroll
        for (int v = 0; v < 8; ++v) {
            float p = 0.f;
#pragma unroll
            for (int nt = 0; nt < 6; ++nt) {
                float hv = acc[nt][v] + cwc[nt];   // D layout: row v+8h, col m
                p += fmaxf(hv, 0.f) * w2c[nt];
            }
            p += __shfl_xor(p, 1);
            p += __shfl_xor(p, 2);
            p += __shfl_xor(p, 4);
            p += __shfl_xor(p, 8);
            if (m == 0) s_lds[v + (h << 3)][c] = fmaxf(p + b2v, 0.f);
        }
    }
    __syncthreads();

    if (L < 16) {   // softmax over C per row; store f16
        float sv[16];
        float mx = -1e30f;
        for (int c = 0; c < 16; ++c) { sv[c] = s_lds[L][c]; mx = fmaxf(mx, sv[c]); }
        float sum = 0.f;
        for (int c = 0; c < 16; ++c) { sv[c] = __expf(sv[c] - mx); sum += sv[c]; }
        float inv = 1.f / sum;
        _Float16* dst = simiH + (size_t)(r0 + L) * C_;
        for (int c = 0; c < 16; ++c) dst[c] = (_Float16)(sv[c] * inv);
    }
}

// ---------------------------------------------------------------------------
// Kernel 4: per (b,n): updated[c][s] = (1/K) * simi_g^T(16x32) @ x_g(32x96)
// Row gather of x via the Tensor Data Mover (gather mode) when available.
// One wave32 per (b,n); 6 WMMAs each; grid = 16000.
// ---------------------------------------------------------------------------
__global__ void k_einsum(const float* __restrict__ x,
                         const int* __restrict__ adj,
                         const _Float16* __restrict__ simiH,
                         float* __restrict__ out) {
    __shared__ float    xf[32][S_];    // gathered x rows (f32, TDM target)
    __shared__ _Float16 sh[32][C_];    // gathered simi rows (f16)
    const int L = threadIdx.x;
    const int h = L >> 4;
    const int m = L & 15;
    const int r = blockIdx.x;          // flat b*N + n
    const int b = r / N_;
    const int j = adj[(size_t)r * K_ + L];
    const float* xb = x + (size_t)b * N_ * S_;

#if defined(__AMDGCN__) && __has_builtin(__builtin_amdgcn_tensor_load_to_lds)
    // ---- TDM gather: 2 descriptors, 16 rows each, tile_dim0 = 96 f32 ----
    {
        unsigned long long gax = (unsigned long long)(size_t)xb;
        unsigned ldsx = (unsigned)(size_t)(__attribute__((address_space(3))) void*)&xf[0][0];
#pragma unroll
        for (int part = 0; part < 2; ++part) {
            v4u g0;
            g0[0] = 1u | (1u << 31);                       // count=1, gather_mode=1, 16b idx
            g0[1] = ldsx + (unsigned)part * 16u * 384u;    // lds_addr (bytes)
            g0[2] = (unsigned)(gax & 0xFFFFFFFFull);       // global_addr lo
            g0[3] = (unsigned)((gax >> 32) & 0x1FFFFFFull) | (2u << 30);  // hi + type=2
            v8i g1;
            g1[0] = (int)(2u << 16);     // data_size = 4B
            g1[1] = (int)(96u << 16);    // tensor_dim0 = 96   (bits 48..79)
            g1[2] = (int)(500u << 16);   // tensor_dim1 = 500  (bits 80..111)
            g1[3] = (int)(96u << 16);    // tile_dim0 = 96     (bits 112..127)
            g1[4] = 16;                  // tile_dim1 = #indices = 16
            g1[5] = 96;                  // tensor_dim0_stride (lo 32)
            g1[6] = 0;
            g1[7] = 0;
            v4i g2, g3;
#pragma unroll
            for (int w = 0; w < 4; ++w) {
                int i0 = part * 16 + w * 2;
                unsigned lo = (unsigned)__builtin_amdgcn_readlane(j, i0) & 0xFFFFu;
                unsigned hi = (unsigned)__builtin_amdgcn_readlane(j, i0 + 1) & 0xFFFFu;
                g2[w] = (int)(lo | (hi << 16));
                int i1 = part * 16 + 8 + w * 2;
                unsigned lo2 = (unsigned)__builtin_amdgcn_readlane(j, i1) & 0xFFFFu;
                unsigned hi2 = (unsigned)__builtin_amdgcn_readlane(j, i1 + 1) & 0xFFFFu;
                g3[w] = (int)(lo2 | (hi2 << 16));
            }
#if __clang_major__ >= 23
            v8i g4;
#pragma unroll
            for (int w = 0; w < 8; ++w) g4[w] = 0;
            __builtin_amdgcn_tensor_load_to_lds(g0, g1, g2, g3, g4, 0);
#else
            __builtin_amdgcn_tensor_load_to_lds(g0, g1, g2, g3, 0);
#endif
        }
    }
    // simi rows are tiny (32B each): gather manually
    for (int kk = 0; kk < 32; ++kk) {
        int jj = __shfl(j, kk);
        if (L < 16) sh[kk][L] = simiH[((size_t)b * N_ + jj) * C_ + L];
    }
#if __has_builtin(__builtin_amdgcn_s_wait_tensorcnt)
    __builtin_amdgcn_s_wait_tensorcnt(0);
#else
    asm volatile("s_wait_tensorcnt 0x0" ::: "memory");
#endif
    __syncthreads();
#else
    // ---- fallback: cooperative manual gather ----
    for (int kk = 0; kk < 32; ++kk) {
        int jj = __shfl(j, kk);
        const float* xr = xb + (size_t)jj * S_;
#pragma unroll
        for (int u = 0; u < 3; ++u)
            xf[kk][u * 32 + L] = xr[u * 32 + L];
        if (L < 16) sh[kk][L] = simiH[((size_t)b * N_ + jj) * C_ + L];
    }
    __syncthreads();
#endif

    // A fragment: 16x32, row = c (= lane&15), contraction = gathered-k index
    v16h A;
#pragma unroll
    for (int e = 0; e < 16; ++e) {
        int v = e >> 1, jb = e & 1;
        int kk = (h << 4) + e;
        A[e] = sh[kk][m];
    }

    float* orow = out + (size_t)r * C_ * S_;
#pragma unroll
    for (int st = 0; st < 6; ++st) {
        const int scol = st * 16 + m;
        v16h Bf;
#pragma unroll
        for (int e = 0; e < 16; ++e)
            Bf[e] = (_Float16)xf[(h << 4) + e][scol];   // B 32x16 layout
        v8f accv;
#pragma unroll
        for (int i = 0; i < 8; ++i) accv[i] = 0.f;
        accv = __builtin_amdgcn_wmma_f32_16x16x32_f16(
                   false, A, false, Bf, (short)0, accv, false, false);
#pragma unroll
        for (int v = 0; v < 8; ++v) {
            int crow = v + (h << 3);
            orow[(size_t)crow * S_ + scol] = accv[v] * (1.f / (float)K_);
        }
    }
}

// ---------------------------------------------------------------------------
// Kernel 5: centSum[c][s] = sum over all (b,n) of updated_input
// ---------------------------------------------------------------------------
__global__ void k_csum(const float* __restrict__ out, float* __restrict__ centSum) {
    int t = blockIdx.x * blockDim.x + threadIdx.x;
    if (t >= C_ * S_) return;
    float acc = 0.f;
    const float* p = out + t;
    for (int r = 0; r < ROWS_; ++r)
        acc += p[(size_t)r * (C_ * S_)];
    centSum[t] = acc;
}

// ---------------------------------------------------------------------------
// Kernel 6: EMA centroid update + pairwise hinge loss (single block)
// ---------------------------------------------------------------------------
__global__ void k_loss(const float* __restrict__ cent,
                       const float* __restrict__ centSum,
                       float* __restrict__ lossOut) {
    __shared__ float nc[C_][D_];
    __shared__ float red[256];
    const int t = threadIdx.x;
    for (int i = t; i < C_ * D_; i += 256) {
        float mean = centSum[i] * (1.f / (float)ROWS_);
        nc[i / D_][i % D_] = 0.99f * cent[i] + 0.01f * mean;
    }
    __syncthreads();
    const int i = t >> 4, jc = t & 15;   // all 16x16 pairs
    float d2 = 0.f;
    for (int d = 0; d < D_; ++d) {
        float df = nc[i][d] - nc[jc][d];   // mean-adjustment cancels in diff
        d2 += df * df;
    }
    float dist = sqrtf(fmaxf(d2, 1e-30f));
    float contrib = 0.f;
    if (i != jc) {
        float v = fmaxf(0.5f - dist, 0.f);
        contrib = v * v;
    }
    red[t] = contrib;
    __syncthreads();
    for (int off = 128; off > 0; off >>= 1) {
        if (t < off) red[t] += red[t + off];
        __syncthreads();
    }
    if (t == 0) lossOut[0] = red[0];
}

// ---------------------------------------------------------------------------
extern "C" void kernel_launch(void* const* d_in, const int* in_sizes, int n_in,
                              void* d_out, int out_size, void* d_ws, size_t ws_size,
                              hipStream_t stream) {
    const float* f     = (const float*)d_in[0];  // fushed_features (B,N,D)
    const float* x     = (const float*)d_in[1];  // input_data (B,1,N,S)
    const float* cent  = (const float*)d_in[2];  // centroids (C,D)
    const float* w1    = (const float*)d_in[3];  // (2D,D)
    const float* b1    = (const float*)d_in[4];  // (D,)
    const float* w2    = (const float*)d_in[5];  // (D,1)
    const float* b2    = (const float*)d_in[6];  // (1,)
    const float* gamma = (const float*)d_in[7];  // (N,)
    const float* beta  = (const float*)d_in[8];  // (N,)
    const int*   adj   = (const int*)d_in[9];    // (B,N,K)

    float* out = (float*)d_out;                  // updated_input (B,N,C,S) + loss

    // workspace layout (f32 units)
    float* ws      = (float*)d_ws;
    float* aS      = ws;                          // 500
    float* cS      = ws + 512;                    // 500
    float* cw      = ws + 1024;                   // 1536
    float* centSum = ws + 2560;                   // 1536
    _Float16* simiH = (_Float16*)(ws + 4096);     // B*N*C f16 = 512000 B
    _Float16* w1t   = (_Float16*)(ws + 132096);   // 9216 f16 = 18432 B

    k_bn_stats<<<N_, 128, 0, stream>>>(f, gamma, beta, aS, cS);
    k_prep<<<6, 256, 0, stream>>>(cent, w1, b1, cw, w1t);
    k_score<<<ROWS_ / 16, 32, 0, stream>>>(f, w1t, w2, b2, aS, cS, cw, simiH);
    k_einsum<<<ROWS_, 32, 0, stream>>>(x, adj, simiH, out);
    k_csum<<<6, 256, 0, stream>>>(out, centSum);
    k_loss<<<1, 256, 0, stream>>>(cent, centSum, out + (size_t)ROWS_ * C_ * S_);
}
